// GAT_33938831573041
// MI455X (gfx1250) — matrix operand order repaired and checked by reading, not compiled, not run on hardware
//
#include <hip/hip_runtime.h>

#define NODES 50000
#define NEDGES 800000

typedef unsigned short ushort_t;
typedef __attribute__((ext_vector_type(16))) __bf16 v16bf;
typedef __attribute__((ext_vector_type(8)))  float  v8f;

__device__ inline ushort_t f2bf(float f) {
    unsigned int u = __float_as_uint(f);
    u += 0x7FFFu + ((u >> 16) & 1u);   // round-to-nearest-even
    return (ushort_t)(u >> 16);
}

__device__ inline void atomicMaxF(float* addr, float val) {
    unsigned int* ua = (unsigned int*)addr;
    unsigned int old = *ua;
    while (__uint_as_float(old) < val) {
        unsigned int assumed = old;
        old = atomicCAS(ua, assumed, __float_as_uint(val));
        if (old == assumed) break;
    }
}

// ---- pack X[N,K] (f32) into per-lane WMMA A-fragment order (bf16) ----
// fragment idx = ((mtile*Ksteps + kstep)*32 + lane), 16 bf16 per fragment.
// lane: row = mtile*16 + lane%16, h = lane/16;
// elems 0..7  <- X[row, kbase + 8h .. +7]
// elems 8..15 <- X[row, kbase + 16 + 8h .. +7]
__global__ void pack_a_kernel(const float* __restrict__ X, ushort_t* __restrict__ Ap,
                              int K, int Mtiles, int Ksteps) {
    int idx = blockIdx.x * blockDim.x + threadIdx.x;
    int total = Mtiles * Ksteps * 32;
    if (idx >= total) return;
    int lane  = idx & 31;
    int rest  = idx >> 5;
    int kstep = rest % Ksteps;
    int mtile = rest / Ksteps;
    int row   = mtile * 16 + (lane & 15);
    int h     = lane >> 4;
    int kbase = kstep * 32;
    ushort_t tmp[16];
    const float* x0 = X + (size_t)row * K + kbase + 8 * h;
    #pragma unroll
    for (int i = 0; i < 8; ++i) tmp[i] = f2bf(x0[i]);
    const float* x1 = X + (size_t)row * K + kbase + 16 + 8 * h;
    #pragma unroll
    for (int i = 0; i < 8; ++i) tmp[8 + i] = f2bf(x1[i]);
    uint4* dst = (uint4*)(Ap + (size_t)idx * 16);
    uint4* s   = (uint4*)tmp;
    dst[0] = s[0];
    dst[1] = s[1];
}

// ---- pack W[K,Nout] (f32) into WMMA B-fragment order, [kstep][ntile] major ----
// fragment idx = ((kstep*Ntiles + ntile)*32 + lane): for a fixed kstep the
// Ntiles fragments are contiguous -> immediate-offset loads in the GEMM.
// lane: col = ntile*16 + lane%16, h = lane/16; elems i<8: K = kbase + 8h + i, i>=8: K = kbase + 16 + 8h + (i-8)
__global__ void pack_w_kernel(const float* __restrict__ W, ushort_t* __restrict__ Bp,
                              int K, int Nout, int Ntiles, int Ksteps) {
    int idx = blockIdx.x * blockDim.x + threadIdx.x;
    int total = Ntiles * Ksteps * 32;
    if (idx >= total) return;
    int lane  = idx & 31;
    int rest  = idx >> 5;
    int ntile = rest % Ntiles;
    int kstep = rest / Ntiles;
    int col   = ntile * 16 + (lane & 15);
    int h     = lane >> 4;
    int kbase = kstep * 32;
    ushort_t tmp[16];
    #pragma unroll
    for (int i = 0; i < 16; ++i)
        tmp[i] = f2bf(W[(size_t)(kbase + 8 * h + i + (i >= 8 ? 8 : 0)) * Nout + col]);
    uint4* dst = (uint4*)(Bp + (size_t)idx * 16);
    uint4* s   = (uint4*)tmp;
    dst[0] = s[0];
    dst[1] = s[1];
}

// ---- WMMA GEMM, register-blocked over NB N-tiles per wave, software-pipelined.
// Double-buffers the A fragment and all NB B fragments: K-step ks+1's loads are
// issued before the WMMA chain of step ks, so waits are against loads launched
// one full chain earlier instead of load->wait->wmma serialization.
template<int NB, int KSTEPS, int NTILES>
__global__ void gemm_wmma_kernel(const ushort_t* __restrict__ Ap,
                                 const ushort_t* __restrict__ Bp,
                                 float* __restrict__ C,
                                 int Mtiles, int Ngroups) {
    constexpr int NOUT = NTILES * 16;
    int wave = (int)((blockIdx.x * blockDim.x + threadIdx.x) >> 5);
    int lane = threadIdx.x & 31;
    if (wave >= Mtiles * Ngroups) return;   // uniform per wave -> EXEC all-1s inside
    int mtile = wave / Ngroups;
    int ngrp  = wave % Ngroups;
    v8f acc[NB];
    #pragma unroll
    for (int i = 0; i < NB; ++i) acc[i] = (v8f){};
    const v16bf* A = (const v16bf*)Ap + (size_t)(mtile * KSTEPS) * 32 + lane;
    const v16bf* B = (const v16bf*)Bp + (size_t)(ngrp * NB) * 32 + lane;

    v16bf a_cur = A[0];
    v16bf b_cur[NB];
    #pragma unroll
    for (int nb = 0; nb < NB; ++nb) b_cur[nb] = B[nb * 32];

    #pragma unroll
    for (int ks = 0; ks < KSTEPS; ++ks) {
        v16bf a_nxt;
        v16bf b_nxt[NB];
        if (ks + 1 < KSTEPS) {
            a_nxt = A[(ks + 1) * 32];
            #pragma unroll
            for (int nb = 0; nb < NB; ++nb)
                b_nxt[nb] = B[((ks + 1) * NTILES + nb) * 32];
        }
        #pragma unroll
        for (int nb = 0; nb < NB; ++nb)
            acc[nb] = __builtin_amdgcn_wmma_f32_16x16x32_bf16(false, a_cur, false, b_cur[nb],
                                                              (short)0, acc[nb],
                                                              false, false);
        a_cur = a_nxt;
        #pragma unroll
        for (int nb = 0; nb < NB; ++nb) b_cur[nb] = b_nxt[nb];
    }

    // C layout: lane col = lane%16; rows = mtile*16 + 8*(lane/16) + r
    int colbase = ngrp * NB * 16 + (lane & 15);
    int rbase   = mtile * 16 + 8 * (lane >> 4);
    float* Cb = C + (size_t)rbase * NOUT + colbase;
    #pragma unroll
    for (int nb = 0; nb < NB; ++nb) {
        #pragma unroll
        for (int r = 0; r < 8; ++r)
            Cb[r * NOUT + nb * 16] = acc[nb][r];
    }
}

// ---- per-node attention logits: al_s/al_d[n,h] = <h[n,h,:], a_src/dst[h,:]> ----
__global__ void attn_logits_kernel(const float* __restrict__ Hf,
                                   const float* __restrict__ asrc,
                                   const float* __restrict__ adst,
                                   float* __restrict__ als, float* __restrict__ ald,
                                   int H, int C) {
    int idx = blockIdx.x * blockDim.x + threadIdx.x;
    if (idx >= NODES * H) return;
    int n = idx / H, hh = idx % H;
    const float* hrow = Hf + (size_t)n * H * C + hh * C;
    const float* as = asrc + hh * C;
    const float* ad = adst + hh * C;
    float s = 0.f, d = 0.f;
    for (int c = 0; c < C; ++c) { s += hrow[c] * as[c]; d += hrow[c] * ad[c]; }
    als[idx] = s;
    ald[idx] = d;
}

__global__ void init_kernel(float* __restrict__ p, int n, float v) {
    int idx = blockIdx.x * blockDim.x + threadIdx.x;
    if (idx < n) p[idx] = v;
}

__device__ inline void edge_decode(const int* __restrict__ eidx, int e, int& src, int& dst) {
    if (e < NEDGES) { src = eidx[e]; dst = eidx[NEDGES + e]; }
    else            { src = dst = e - NEDGES; }
}

__device__ inline float edge_logit(const float* __restrict__ als, const float* __restrict__ ald,
                                   int src, int dst, int hh, int H) {
    float l = als[src * H + hh] + ald[dst * H + hh];
    return l > 0.f ? l : 0.2f * l;   // leaky_relu(0.2)
}

__global__ void edge_max_kernel(const int* __restrict__ eidx,
                                const float* __restrict__ als, const float* __restrict__ ald,
                                float* __restrict__ m, int H) {
    int idx = blockIdx.x * blockDim.x + threadIdx.x;
    int total = (NEDGES + NODES) * H;
    if (idx >= total) return;
    int e = idx / H, hh = idx % H, src, dst;
    edge_decode(eidx, e, src, dst);
    atomicMaxF(&m[dst * H + hh], edge_logit(als, ald, src, dst, hh, H));
}

__global__ void edge_denom_kernel(const int* __restrict__ eidx,
                                  const float* __restrict__ als, const float* __restrict__ ald,
                                  const float* __restrict__ m, float* __restrict__ denom, int H) {
    int idx = blockIdx.x * blockDim.x + threadIdx.x;
    int total = (NEDGES + NODES) * H;
    if (idx >= total) return;
    int e = idx / H, hh = idx % H, src, dst;
    edge_decode(eidx, e, src, dst);
    float l = edge_logit(als, ald, src, dst, hh, H);
    atomicAdd(&denom[dst * H + hh], __expf(l - m[dst * H + hh]));
}

__global__ void edge_agg_kernel(const int* __restrict__ eidx,
                                const float* __restrict__ als, const float* __restrict__ ald,
                                const float* __restrict__ m, const float* __restrict__ denom,
                                const float* __restrict__ Hf, float* __restrict__ out,
                                int H, int C) {
    int idx = blockIdx.x * blockDim.x + threadIdx.x;
    int total = (NEDGES + NODES) * H;
    if (idx >= total) return;
    int e = idx / H, hh = idx % H, src, dst;
    edge_decode(eidx, e, src, dst);
    float l = edge_logit(als, ald, src, dst, hh, H);
    float w = __expf(l - m[dst * H + hh]) / denom[dst * H + hh];
    const float* hs = Hf + (size_t)src * H * C + hh * C;
    float* od = out + (size_t)dst * H * C + hh * C;
    for (int c = 0; c < C; ++c)
        atomicAdd(&od[c], hs[c] * w);
}

__global__ void bias_relu_kernel(float* __restrict__ out, const float* __restrict__ b,
                                 int Nout, int relu) {
    int idx = blockIdx.x * blockDim.x + threadIdx.x;
    if (idx >= NODES * Nout) return;
    float v = out[idx] + b[idx % Nout];
    out[idx] = relu ? fmaxf(v, 0.f) : v;
}

static void run_attention(const float* Hfeat, int Nout, int H, int C,
                          const float* asrc, const float* adst, const float* b,
                          float* outbuf, int relu, const int* eidx,
                          float* als, float* ald, float* m, float* denom,
                          hipStream_t stream) {
    int nh = NODES * H;
    attn_logits_kernel<<<(nh + 255) / 256, 256, 0, stream>>>(Hfeat, asrc, adst, als, ald, H, C);
    init_kernel<<<(nh + 255) / 256, 256, 0, stream>>>(m, nh, -3.0e38f);
    init_kernel<<<(nh + 255) / 256, 256, 0, stream>>>(denom, nh, 0.f);

    int totE = (NEDGES + NODES) * H;
    edge_max_kernel<<<(totE + 255) / 256, 256, 0, stream>>>(eidx, als, ald, m, H);
    edge_denom_kernel<<<(totE + 255) / 256, 256, 0, stream>>>(eidx, als, ald, m, denom, H);

    int no = NODES * Nout;
    init_kernel<<<(no + 255) / 256, 256, 0, stream>>>(outbuf, no, 0.f);
    edge_agg_kernel<<<(totE + 255) / 256, 256, 0, stream>>>(eidx, als, ald, m, denom,
                                                            Hfeat, outbuf, H, C);
    bias_relu_kernel<<<(no + 255) / 256, 256, 0, stream>>>(outbuf, b, Nout, relu);
}

template<int NB, int KSTEPS, int NTILES>
static void run_gemm(const float* X, ushort_t* Apack, ushort_t* Wpack,
                     const float* W, float* Hfeat, hipStream_t stream) {
    constexpr int K = KSTEPS * 32;
    constexpr int NOUT = NTILES * 16;
    int Mtiles = NODES / 16;
    int Ngroups = NTILES / NB;

    int tA = Mtiles * KSTEPS * 32;
    pack_a_kernel<<<(tA + 255) / 256, 256, 0, stream>>>(X, Apack, K, Mtiles, KSTEPS);
    int tW = NTILES * KSTEPS * 32;
    pack_w_kernel<<<(tW + 255) / 256, 256, 0, stream>>>(W, Wpack, K, NOUT, NTILES, KSTEPS);

    int waves = Mtiles * Ngroups;
    gemm_wmma_kernel<NB, KSTEPS, NTILES><<<(waves + 7) / 8, 256, 0, stream>>>(
        Apack, Wpack, Hfeat, Mtiles, Ngroups);
}

extern "C" void kernel_launch(void* const* d_in, const int* in_sizes, int n_in,
                              void* d_out, int out_size, void* d_ws, size_t ws_size,
                              hipStream_t stream) {
    const float* x    = (const float*)d_in[0];
    const int*   eidx = (const int*)  d_in[1];
    const float* W1   = (const float*)d_in[2];
    const float* a1s  = (const float*)d_in[3];
    const float* a1d  = (const float*)d_in[4];
    const float* b1   = (const float*)d_in[5];
    const float* W2   = (const float*)d_in[6];
    const float* a2s  = (const float*)d_in[7];
    const float* a2d  = (const float*)d_in[8];
    const float* b2   = (const float*)d_in[9];
    const float* W3   = (const float*)d_in[10];
    const float* a3s  = (const float*)d_in[11];
    const float* a3d  = (const float*)d_in[12];
    const float* b3   = (const float*)d_in[13];

    char* ws = (char*)d_ws;
    size_t off = 0;
    auto carve = [&](size_t bytes) -> char* {
        char* p = ws + off;
        off += (bytes + 255) & ~(size_t)255;
        return p;
    };
    ushort_t* Apack = (ushort_t*)carve((size_t)NODES * 224 * 2);   // max K=224
    ushort_t* Wpack = (ushort_t*)carve(131072);                    // max 192*240 bf16
    float*    Hfeat = (float*)   carve((size_t)NODES * 240 * 4);   // max Nout=240
    float*    OUT   = (float*)   carve((size_t)NODES * 224 * 4);   // max agg out (L1/L2)
    float*    als   = (float*)   carve((size_t)NODES * 8 * 4);
    float*    ald   = (float*)   carve((size_t)NODES * 8 * 4);
    float*    mbuf  = (float*)   carve((size_t)NODES * 8 * 4);
    float*    dbuf  = (float*)   carve((size_t)NODES * 8 * 4);

    // Layer 1: 128 -> 7x32 (relu); 14 N-tiles = 2 groups of 7, 4 K-steps
    run_gemm<7, 4, 14>(x, Apack, Wpack, W1, Hfeat, stream);
    run_attention(Hfeat, 224, 7, 32, a1s, a1d, b1, OUT, 1, eidx,
                  als, ald, mbuf, dbuf, stream);
    // Layer 2: 224 -> 6x32 (relu); 12 N-tiles = 2 groups of 6, 7 K-steps
    run_gemm<6, 7, 12>(OUT, Apack, Wpack, W2, Hfeat, stream);
    run_attention(Hfeat, 192, 6, 32, a2s, a2d, b2, OUT, 1, eidx,
                  als, ald, mbuf, dbuf, stream);
    // Layer 3: 192 -> 6x40 (no relu); 15 N-tiles = 3 groups of 5, 6 K-steps
    run_gemm<5, 6, 15>(OUT, Apack, Wpack, W3, Hfeat, stream);
    run_attention(Hfeat, 240, 6, 40, a3s, a3d, b3, (float*)d_out, 0, eidx,
                  als, ald, mbuf, dbuf, stream);
}
